// Attribute_Decoder_87282325390065
// MI455X (gfx1250) — matrix operand order repaired and checked by reading, not compiled, not run on hardware
//
#include <hip/hip_runtime.h>
#include <hip/hip_bf16.h>
#include <stdint.h>

typedef __attribute__((ext_vector_type(16))) _Float16 v16h;
typedef __attribute__((ext_vector_type(8)))  _Float16 v8h;
typedef __attribute__((ext_vector_type(8)))  float    v8f;

#define F_IN 128
#define KTOT 256            // concat(agg, x) -> K = 256
#define NKB  8              // 256 / 32 K-blocks per WMMA chain
#define NCT  8              // 128 / 16 output column tiles
#define LDS_STRIDE 272      // 256 + 16 halves pad -> 544B row stride (32B aligned)

// ---------------------------------------------------------------- zero
__global__ void zero_kernel(float* __restrict__ p, int n) {
    int i = blockIdx.x * blockDim.x + threadIdx.x;
    if (i < n) p[i] = 0.0f;
}

// ------------------------------------------------- pack B for WMMA lanes
// Bp flat f16 index = ((ct*8 + kb)*32 + lane)*16 + t
//   lane<16  : holds column n = ct*16 + lane,      K = kb*32 + (t&7) + 2*(t&8)      (t=0..15)
//   lane>=16 : holds column n = ct*16 + (lane-16), K = kb*32 + 8 + (t&7) + 2*(t&8)
__global__ void pack_weights_kernel(const float* __restrict__ Wl,
                                    const float* __restrict__ Wr,
                                    _Float16* __restrict__ Bp) {
    int i = blockIdx.x * blockDim.x + threadIdx.x;     // 8*8*32*16 = 32768
    if (i >= NCT * NKB * 32 * 16) return;
    int t    =  i        & 15;
    int lane = (i >> 4)  & 31;
    int kb   = (i >> 9)  & 7;
    int ct   = (i >> 12) & 7;
    int k = kb * 32 + ((lane & 16) >> 1) + (t & 7) + ((t & 8) << 1); // 0..255
    int n = ct * 16 + (lane & 15);                     // 0..127
    float v = (k < F_IN) ? Wl[k * F_IN + n] : Wr[(k - F_IN) * F_IN + n];
    Bp[i] = (_Float16)v;
}

// --------------------------------------------------- edge scatter-add
// one wave32 per edge: contiguous 512B row gather, float4 per lane,
// 4 fp32 atomics per lane into agg[dst], 1 atomic into deg[dst].
__global__ void aggregate_kernel(const float* __restrict__ xin,
                                 const long long* __restrict__ src,
                                 const long long* __restrict__ dst,
                                 float* __restrict__ agg,
                                 float* __restrict__ deg, int nE) {
    int gtid = blockIdx.x * blockDim.x + threadIdx.x;
    int e    = gtid >> 5;
    int lane = gtid & 31;
    if (e >= nE) return;
    long long s = src[e];
    long long d = dst[e];
    float4 v = *((const float4*)(xin + (size_t)s * F_IN) + lane);
    float* ar = agg + (size_t)d * F_IN + lane * 4;
    atomicAdd(ar + 0, v.x);
    atomicAdd(ar + 1, v.y);
    atomicAdd(ar + 2, v.z);
    atomicAdd(ar + 3, v.w);
    if (lane == 0) atomicAdd(deg + d, 1.0f);
}

// ---------------------- mean-normalize + concat -> f16 A matrix [N, 256]
__global__ void build_A_kernel(const float* __restrict__ agg,
                               const float* __restrict__ deg,
                               const float* __restrict__ xin,
                               _Float16* __restrict__ A, int nN) {
    int i = blockIdx.x * blockDim.x + threadIdx.x;     // nN * 128 threads
    if (i >= nN * F_IN) return;
    int n = i >> 7;
    int t = i & 127;
    float dg = fmaxf(deg[n], 1.0f);
    A[(size_t)n * KTOT + t]        = (_Float16)(agg[i] / dg);
    A[(size_t)n * KTOT + F_IN + t] = (_Float16)xin[i];
}

// ---------------------------------------------- fused WMMA GEMM + bias + relu
// block = 256 threads (8 waves). Block owns 16 rows; wave w owns column tile w.
// A tile (16 x 256 f16) staged in LDS once per block.
__global__ void __launch_bounds__(256)
sage_gemm_kernel(const _Float16* __restrict__ A,
                 const _Float16* __restrict__ Bp,
                 const float* __restrict__ bias,
                 float* __restrict__ out) {
    __shared__ _Float16 As[16][LDS_STRIDE];            // 544B row stride, 32B aligned

    const int row_base = blockIdx.x * 16;
    const int tid = threadIdx.x;

    // cooperative LDS fill: 256 threads x 32B = 16 rows x 512B
    {
        int r = tid >> 4;
        int c = (tid & 15) * 16;                       // half index 0..240
        const v16h* gsrc = (const v16h*)(A + (size_t)(row_base + r) * KTOT + c);
        *(v16h*)(&As[r][c]) = *gsrc;
    }
    __syncthreads();

    const int wv   = tid >> 5;                         // column tile 0..7
    const int lane = tid & 31;
    const int m    = lane & 15;
    const int ko   = (lane & 16) >> 1;                 // 0 (lanes<16) or 8

    // B pointer in v16h units: ((ct*8 + kb)*32 + lane)
    const v16h* bpt = (const v16h*)Bp + (size_t)(wv * NKB) * 32 + lane;

    v8f c = {};
    union AV { v16h v; v8h h[2]; };
#pragma unroll
    for (int kb = 0; kb < NKB; ++kb) {
        // ISA 16-bit A layout: VGPRs 0-3 = K base+0..7, VGPRs 4-7 = K base+16..23,
        // where base = 8 for lanes 16-31.
        const _Float16* ap = &As[m][kb * 32 + ko];
        AV a;
        a.h[0] = *(const v8h*)(ap);                    // K = kb*32 + ko + 0..7
        a.h[1] = *(const v8h*)(ap + 16);               // K = kb*32 + ko + 16..23
        v16h b = bpt[kb * 32];                         // coalesced 32B/lane
        c = __builtin_amdgcn_wmma_f32_16x16x32_f16(
                /*neg_a=*/false, a.v, /*neg_b=*/false, b,
                /*c_mod=*/(short)0, c, /*reuse_a=*/false, /*reuse_b=*/false);
    }

    // epilogue: C/D layout VGPR r -> M = r (+8 for lanes>=16), N = lane&15
    const int col   = wv * 16 + (lane & 15);
    const int mbase = (lane & 16) ? 8 : 0;
    const float bv  = bias[col];
#pragma unroll
    for (int r = 0; r < 8; ++r) {
        float v = c[r] + bv;
        v = v > 0.0f ? v : 0.0f;
        out[(size_t)(row_base + mbase + r) * F_IN + col] = v;
    }
}

// ------------------------------------------------------------------ launch
static inline size_t align256(size_t x) { return (x + 255) & ~(size_t)255; }

extern "C" void kernel_launch(void* const* d_in, const int* in_sizes, int n_in,
                              void* d_out, int out_size, void* d_ws, size_t ws_size,
                              hipStream_t stream) {
    const float*     x    = (const float*)d_in[0];
    const long long* ei   = (const long long*)d_in[1];   // int64 per reference
    const float*     W1l  = (const float*)d_in[2];
    const float*     b1   = (const float*)d_in[3];
    const float*     W1r  = (const float*)d_in[4];
    const float*     W2l  = (const float*)d_in[5];
    const float*     b2   = (const float*)d_in[6];
    const float*     W2r  = (const float*)d_in[7];
    float*           outp = (float*)d_out;

    const int N = in_sizes[0] / F_IN;                    // 50000
    const int E = in_sizes[1] / 2;                       // 600000
    const long long* src = ei;
    const long long* dst = ei + E;

    // workspace layout
    char* ws = (char*)d_ws;
    size_t off = 0;
    float*     agg = (float*)(ws + off);     off = align256(off + (size_t)N * F_IN * 4 + (size_t)N * 4);
    float*     deg = agg + (size_t)N * F_IN; // deg lives right after agg (contiguous -> one zero pass)
    _Float16*  A   = (_Float16*)(ws + off);  off = align256(off + (size_t)N * KTOT * 2);
    float*     h   = (float*)(ws + off);     off = align256(off + (size_t)N * F_IN * 4);
    _Float16*  Bp  = (_Float16*)(ws + off);  off = align256(off + (size_t)NCT * NKB * 32 * 16 * 2);

    const int zeroN   = N * F_IN + N;
    const int zeroBlk = (zeroN + 255) / 256;
    const int packBlk = (NCT * NKB * 32 * 16 + 255) / 256;     // 128
    const int aggBlk  = (E * 32 + 255) / 256;                  // one wave per edge
    const int bldBlk  = (N * F_IN + 255) / 256;
    const int gemmBlk = N / 16;                                // 3125 (N % 16 == 0)

    // ---------------- layer 1: F_IN -> HID, relu ----------------
    zero_kernel        <<<zeroBlk, 256, 0, stream>>>(agg, zeroN);
    pack_weights_kernel<<<packBlk, 256, 0, stream>>>(W1l, W1r, Bp);
    aggregate_kernel   <<<aggBlk,  256, 0, stream>>>(x, src, dst, agg, deg, E);
    build_A_kernel     <<<bldBlk,  256, 0, stream>>>(agg, deg, x, A, N);
    sage_gemm_kernel   <<<gemmBlk, 256, 0, stream>>>(A, Bp, b1, h);

    // ---------------- layer 2: HID -> F_IN, relu ----------------
    zero_kernel        <<<zeroBlk, 256, 0, stream>>>(agg, zeroN);
    pack_weights_kernel<<<packBlk, 256, 0, stream>>>(W2l, W2r, Bp);
    aggregate_kernel   <<<aggBlk,  256, 0, stream>>>(h, src, dst, agg, deg, E);
    build_A_kernel     <<<bldBlk,  256, 0, stream>>>(agg, deg, h, A, N);
    sage_gemm_kernel   <<<gemmBlk, 256, 0, stream>>>(A, Bp, b2, outp);
}
